// SelfAttention_52166672778037
// MI455X (gfx1250) — matrix operand order repaired and checked by reading, not compiled, not run on hardware
//
#include <hip/hip_runtime.h>

// Problem constants (match reference)
#define HEADS   16
#define HDIM    64
#define EMBD    1024
#define LSEQ    1024
#define BATCH   4
#define MAXREL  512
#define RELROWS (2 * MAXREL + 1)   // 1025
#define RELPAD  2048               // zero-padded table extent
#define JW      544                // j-window width per 512-key half (17 * 32)

typedef _Float16 h16;
typedef _Float16 v8h  __attribute__((ext_vector_type(8), aligned(16), may_alias));
typedef _Float16 v16h __attribute__((ext_vector_type(16)));
typedef float    v8f  __attribute__((ext_vector_type(8)));

// ---------------------------------------------------------------------------
// D = A(16x32 f16) * B(32x16 f16) + C(16x16 f32)
// ---------------------------------------------------------------------------
__device__ __forceinline__ v8f wmma_f16(v16h a, v16h b, v8f c) {
    return __builtin_amdgcn_wmma_f32_16x16x32_f16(
        /*neg_a=*/false, a, /*neg_b=*/false, b,
        /*c_mod=*/(short)0, c, /*reuse_a=*/false, /*reuse_b=*/false);
}

// A-operand gather: 16 contiguous halves (K = base..+15)
__device__ __forceinline__ v16h mk16_a(const h16* p) {
    v8h lo = *(const v8h*)p;
    v8h hi = *(const v8h*)(p + 8);
    v16h r;
#pragma unroll
    for (int i = 0; i < 8; ++i) { r[i] = lo[i]; r[i + 8] = hi[i]; }
    return r;
}

// B-operand gather: 16 contiguous halves (K ascending)
__device__ __forceinline__ v16h mk16_b(const h16* p) {
    v8h lo = *(const v8h*)p;
    v8h hi = *(const v8h*)(p + 8);
    v16h r;
#pragma unroll
    for (int i = 0; i < 8; ++i) { r[i] = lo[i]; r[i + 8] = hi[i]; }
    return r;
}

// ---------------------------------------------------------------------------
// table prep kernels
// ---------------------------------------------------------------------------
__global__ void cvt_f32_f16(const float* __restrict__ in, h16* __restrict__ out, int n) {
    int i = blockIdx.x * blockDim.x + threadIdx.x;
    if (i < n) out[i] = (h16)in[i];
}

// relK padded: out[2048][64], rows >= 1025 are zero
__global__ void prep_relK(const float* __restrict__ in, h16* __restrict__ out) {
    int i = blockIdx.x * blockDim.x + threadIdx.x;   // over RELPAD*HDIM
    int row = i >> 6;
    out[i] = (row < RELROWS) ? (h16)in[i] : (h16)0.0f;
}

// relV transposed + padded: out[64][2048], cols >= 1025 are zero
__global__ void prep_relVT(const float* __restrict__ in, h16* __restrict__ out) {
    int i = blockIdx.x * blockDim.x + threadIdx.x;   // over HDIM*RELPAD
    int d = i >> 11;
    int j = i & (RELPAD - 1);
    out[i] = (j < RELROWS) ? (h16)in[(size_t)j * HDIM + d] : (h16)0.0f;
}

// ---------------------------------------------------------------------------
// Per-head QKV projection:  y[b,l,h,n] = sum_d x[b,l,h*64+d] * W[n,d] + bias[n]
// blockIdx.z selects q/k/v.  V is stored transposed [b,h,d,l] for the PV GEMM.
// ---------------------------------------------------------------------------
__global__ __launch_bounds__(256) void proj_kernel(
    const float* __restrict__ q_in, const float* __restrict__ k_in,
    const float* __restrict__ v_in,
    const float* __restrict__ Wq, const float* __restrict__ bq,
    const float* __restrict__ Wk, const float* __restrict__ bk,
    const float* __restrict__ Wv, const float* __restrict__ bv,
    h16* __restrict__ qh, h16* __restrict__ kh, h16* __restrict__ vT)
{
    const int wave = threadIdx.x >> 5;
    const int lane = threadIdx.x & 31;
    const int bh   = blockIdx.y;
    const int b    = bh >> 4;
    const int h    = bh & 15;
    const int which = blockIdx.z;
    const int l0   = blockIdx.x * 128 + wave * 16;

    const float* x;  const float* W;  const float* bias;
    if (which == 0)      { x = q_in; W = Wq; bias = bq; }
    else if (which == 1) { x = k_in; W = Wk; bias = bk; }
    else                 { x = v_in; W = Wv; bias = bv; }

    const int mrow  = lane & 15;
    const int abase = (lane < 16) ? 0 : 16;
    const int bkoff = (lane < 16) ? 0 : 16;
    const int hi8   = (lane < 16) ? 0 : 8;

    v8f acc[4];
#pragma unroll
    for (int t = 0; t < 4; ++t) acc[t] = {};

#pragma unroll
    for (int ks = 0; ks < 2; ++ks) {
        const float* xp = x + (size_t)(b * LSEQ + l0 + mrow) * EMBD + h * HDIM + ks * 32 + abase;
        v16h a;
#pragma unroll
        for (int i = 0; i < 8; ++i) { a[i] = (h16)xp[i]; a[i + 8] = (h16)xp[8 + i]; }
#pragma unroll
        for (int t = 0; t < 4; ++t) {
            const float* wp = W + (t * 16 + mrow) * HDIM + ks * 32 + bkoff;
            v16h bm;
#pragma unroll
            for (int i = 0; i < 16; ++i) bm[i] = (h16)wp[i];
            acc[t] = wmma_f16(a, bm, acc[t]);
        }
    }

#pragma unroll
    for (int t = 0; t < 4; ++t) {
        const int n = t * 16 + mrow;
        const float bval = bias[n];
#pragma unroll
        for (int r = 0; r < 8; ++r) {
            const int row = l0 + r + hi8;
            const h16 val = (h16)(acc[t][r] + bval);
            if (which == 2)      vT[((size_t)bh * HDIM + n) * LSEQ + row] = val;
            else if (which == 1) kh[((size_t)bh * LSEQ + row) * HDIM + n] = val;
            else                 qh[((size_t)bh * LSEQ + row) * HDIM + n] = val;
        }
    }
}

// ---------------------------------------------------------------------------
// Attention with Shaw relative positions, one wave / workgroup, 34 KB LDS.
// Keys processed in two 512-wide halves. Per half:
//   [1] Prel[m, j-window] = q . relK[j]          (WMMA, into jbuf)
//   [2] pass A: S = Q*K^T (WMMA) + mask + Prel lookup -> sld, track row max
//   [3] merge max with running stats (flash rescale of O/R/lsum)
//   [4] pass B: P = exp(S-m) in place, row sums, O += P*V (WMMA)
//   [5] build C[m,j] (shifted copy of P + clip buckets) into jbuf
//   [6] R += C @ relV^T                          (WMMA)
// ---------------------------------------------------------------------------
__global__ __launch_bounds__(32) void attn_kernel(
    const h16* __restrict__ qh, const h16* __restrict__ kh,
    const h16* __restrict__ vT, const int* __restrict__ mask,
    const h16* __restrict__ relKp, const h16* __restrict__ relVT,
    h16* __restrict__ X)
{
    __shared__ h16 sld[16 * 512];   // scores / softmax numerators  16 KB
    __shared__ h16 jbuf[16 * JW];   // Prel, then C                 17 KB

    const int lane = threadIdx.x & 31;
    const int h    = blockIdx.y;
    const int b    = blockIdx.z;
    const int bh   = b * HEADS + h;
    const int l0   = blockIdx.x * 16;

    const int mrow  = lane & 15;
    const int abase = (lane < 16) ? 0 : 16;
    const int bkoff = (lane < 16) ? 0 : 16;
    const int hi8   = (lane < 16) ? 0 : 8;

    // q A-operands live entirely in registers (A layout, K=64 in two steps)
    const h16* qp = qh + ((size_t)bh * LSEQ + l0) * HDIM;
    const v16h a0 = mk16_a(qp + mrow * HDIM + abase);
    const v16h a1 = mk16_a(qp + mrow * HDIM + 32 + abase);

    float mglob[8], lsum[8];
    v8f oacc[4], racc[4];
#pragma unroll
    for (int r = 0; r < 8; ++r) { mglob[r] = -3.0e38f; lsum[r] = 0.0f; }
#pragma unroll
    for (int t = 0; t < 4; ++t) { oacc[t] = {}; racc[t] = {}; }

    for (int ha = 0; ha < 2; ++ha) {
        const int kbase = ha * 512;
        // j-window base (16-aligned) for this half
        int dmin = kbase - (l0 + 15);
        dmin = min(max(dmin, -MAXREL), MAXREL);
        const int jb16 = (dmin + MAXREL) & ~15;

        // ---- [1] Prel[m, jb16 + 0..543] = q . relK  (34 j-tiles, WMMA)
        for (int jt = 0; jt < JW / 16; ++jt) {
            v8f c = {};
            const h16* bp0 = relKp + (size_t)(jb16 + jt * 16 + mrow) * HDIM + bkoff;
            c = wmma_f16(a0, mk16_b(bp0), c);
            c = wmma_f16(a1, mk16_b(bp0 + 32), c);
#pragma unroll
            for (int r = 0; r < 8; ++r)
                jbuf[(r + hi8) * JW + jt * 16 + mrow] = (h16)c[r];
        }

        // ---- [2] pass A: scores into sld, track running max
        float pmax[8];
#pragma unroll
        for (int r = 0; r < 8; ++r) pmax[r] = -3.0e38f;

        for (int c0 = 0; c0 < 512; c0 += 32) {
            if (c0 + 32 < 512)
                __builtin_prefetch(kh + ((size_t)bh * LSEQ + kbase + c0 + 32) * HDIM, 0, 3);
            v8f s[2]; s[0] = {}; s[1] = {};
#pragma unroll
            for (int t = 0; t < 2; ++t) {
                const h16* bp = kh + ((size_t)bh * LSEQ + kbase + c0 + t * 16 + mrow) * HDIM + bkoff;
                s[t] = wmma_f16(a0, mk16_b(bp), s[t]);
                s[t] = wmma_f16(a1, mk16_b(bp + 32), s[t]);
            }
#pragma unroll
            for (int t = 0; t < 2; ++t) {
#pragma unroll
                for (int r = 0; r < 8; ++r) {
                    const int M    = r + hi8;
                    const int l    = l0 + M;
                    const int kcol = kbase + c0 + t * 16 + mrow;
                    float sv = s[t][r];
                    if (mask[(size_t)(b * LSEQ + l) * LSEQ + kcol] == 0) sv = -1.0e20f;
                    int dist = kcol - l;
                    dist = min(max(dist, -MAXREL), MAXREL);
                    sv += (float)jbuf[M * JW + dist + MAXREL - jb16];
                    sv *= 0.125f;
                    pmax[r] = fmaxf(pmax[r], sv);
                    sld[M * 512 + c0 + t * 16 + mrow] = (h16)sv;
                }
            }
        }

        // ---- [3] reduce max across 16-lane row group, merge + rescale
#pragma unroll
        for (int r = 0; r < 8; ++r) {
            float cm = pmax[r];
#pragma unroll
            for (int off = 1; off < 16; off <<= 1)
                cm = fmaxf(cm, __shfl_xor(cm, off, 32));
            const float mnew  = fmaxf(mglob[r], cm);
            const float scale = __expf(mglob[r] - mnew);
            mglob[r] = mnew;
            lsum[r] *= scale;
#pragma unroll
            for (int t = 0; t < 4; ++t) { oacc[t][r] *= scale; racc[t][r] *= scale; }
        }

        // ---- [4] pass B: exp in place + row sums + PV GEMM
        float psum[8];
#pragma unroll
        for (int r = 0; r < 8; ++r) psum[r] = 0.0f;

        for (int c0 = 0; c0 < 512; c0 += 32) {
#pragma unroll
            for (int t = 0; t < 2; ++t) {
#pragma unroll
                for (int r = 0; r < 8; ++r) {
                    const int idx = (r + hi8) * 512 + c0 + t * 16 + mrow;
                    const float p = __expf((float)sld[idx] - mglob[r]);
                    psum[r] += p;
                    sld[idx] = (h16)p;
                }
            }
            v16h ap = mk16_a(sld + mrow * 512 + c0 + abase);
#pragma unroll
            for (int t = 0; t < 4; ++t) {
                const h16* vp = vT + ((size_t)bh * HDIM + t * 16 + mrow) * LSEQ
                              + kbase + c0 + bkoff;
                oacc[t] = wmma_f16(ap, mk16_b(vp), oacc[t]);
            }
        }
#pragma unroll
        for (int r = 0; r < 8; ++r) {
            float ps = psum[r];
#pragma unroll
            for (int off = 1; off < 16; off <<= 1)
                ps += __shfl_xor(ps, off, 32);
            lsum[r] += ps;
        }

        // ---- [5] build C[m,j] in jbuf: zero, shifted copy of P, clip buckets
        {
            v8h z = {};
            v8h* jb8 = (v8h*)jbuf;
#pragma unroll
            for (int i = 0; i < (16 * JW / 8) / 32; ++i)   // 34 stores/lane
                jb8[lane + 32 * i] = z;
        }
        if (lane < 16) {
            const int m = lane;
            const int l = l0 + m;
            float lowa = 0.0f, higha = 0.0f;
            for (int k = 0; k < 512; ++k) {
                const int dist = kbase + k - l;
                const float p = (float)sld[m * 512 + k];
                if (dist <= -MAXREL)      lowa += p;    // all clip to j = 0
                else if (dist >= MAXREL)  higha += p;   // all clip to j = 1024
                else jbuf[m * JW + dist + MAXREL - jb16] = (h16)p;  // unique j
            }
            if (lowa != 0.0f) {
                const int c = 0 - jb16;                  // valid only if jb16==0
                jbuf[m * JW + c] = (h16)((float)jbuf[m * JW + c] + lowa);
            }
            if (higha != 0.0f) {
                const int c = 2 * MAXREL - jb16;         // j = 1024
                jbuf[m * JW + c] = (h16)((float)jbuf[m * JW + c] + higha);
            }
        }

        // ---- [6] R += C @ relV^T over the j-window (17 K-chunks, WMMA)
        for (int ks = 0; ks < JW / 32; ++ks) {
            v16h ac = mk16_a(jbuf + mrow * JW + ks * 32 + abase);
#pragma unroll
            for (int t = 0; t < 4; ++t) {
                const h16* bp = relVT + (size_t)(t * 16 + mrow) * RELPAD
                              + jb16 + ks * 32 + bkoff;
                racc[t] = wmma_f16(ac, mk16_b(bp), racc[t]);
            }
        }
    }

    // ---- epilogue: normalize and store f16 FC input X[b,l,h*64+d]
#pragma unroll
    for (int r = 0; r < 8; ++r) {
        const float inv = 1.0f / lsum[r];
        const int l = l0 + r + hi8;
#pragma unroll
        for (int t = 0; t < 4; ++t) {
            const int d = t * 16 + mrow;
            X[(size_t)(b * LSEQ + l) * EMBD + h * HDIM + d] =
                (h16)((oacc[t][r] + racc[t][r]) * inv);
        }
    }
}

// ---------------------------------------------------------------------------
// Final FC: out[m,n] = X[m,:] . Wfc[n,:] + bfc[n]   (M=4096, N=K=1024)
// ---------------------------------------------------------------------------
__global__ __launch_bounds__(256) void fc_kernel(
    const h16* __restrict__ X, const h16* __restrict__ Wfc16,
    const float* __restrict__ bfc, float* __restrict__ out)
{
    const int wave = threadIdx.x >> 5;
    const int lane = threadIdx.x & 31;
    const int m0   = (blockIdx.x * 8 + wave) * 16;
    const int c0   = blockIdx.y * 64;

    const int mrow  = lane & 15;
    const int abase = (lane < 16) ? 0 : 16;
    const int bkoff = (lane < 16) ? 0 : 16;
    const int hi8   = (lane < 16) ? 0 : 8;

    v8f acc[4];
#pragma unroll
    for (int t = 0; t < 4; ++t) acc[t] = {};

    for (int ks = 0; ks < 32; ++ks) {
        v16h a = mk16_a(X + (size_t)(m0 + mrow) * EMBD + ks * 32 + abase);
#pragma unroll
        for (int t = 0; t < 4; ++t) {
            const h16* bp = Wfc16 + (size_t)(c0 + t * 16 + mrow) * EMBD + ks * 32 + bkoff;
            acc[t] = wmma_f16(a, mk16_b(bp), acc[t]);
        }
    }

#pragma unroll
    for (int t = 0; t < 4; ++t) {
        const int n = c0 + t * 16 + mrow;
        const float bv = bfc[n];
#pragma unroll
        for (int r = 0; r < 8; ++r)
            out[(size_t)(m0 + r + hi8) * EMBD + n] = acc[t][r] + bv;
    }
}

// ---------------------------------------------------------------------------
extern "C" void kernel_launch(void* const* d_in, const int* in_sizes, int n_in,
                              void* d_out, int out_size, void* d_ws, size_t ws_size,
                              hipStream_t stream)
{
    (void)in_sizes; (void)n_in; (void)out_size; (void)ws_size;

    const float* query = (const float*)d_in[0];
    const float* key_  = (const float*)d_in[1];
    const float* value = (const float*)d_in[2];
    const int*   mask  = (const int*)  d_in[3];
    const float* Wq    = (const float*)d_in[4];
    const float* bq    = (const float*)d_in[5];
    const float* Wk    = (const float*)d_in[6];
    const float* bk    = (const float*)d_in[7];
    const float* Wv    = (const float*)d_in[8];
    const float* bv    = (const float*)d_in[9];
    const float* Wfc   = (const float*)d_in[10];
    const float* bfc   = (const float*)d_in[11];
    const float* relKf = (const float*)d_in[12];
    const float* relVf = (const float*)d_in[13];
    float* out = (float*)d_out;

    // Workspace layout (bytes). Total ~34.5 MB.
    char* ws = (char*)d_ws;
    h16* qh    = (h16*)(ws + 0);
    h16* kh    = (h16*)(ws +  8ull * 1024 * 1024);
    h16* vT    = (h16*)(ws + 16ull * 1024 * 1024);
    h16* X     = (h16*)(ws + 24ull * 1024 * 1024);
    h16* wfc16 = (h16*)(ws + 32ull * 1024 * 1024);
    h16* relKp = (h16*)(ws + 34ull * 1024 * 1024);                  // [2048][64]
    h16* relVT = (h16*)(ws + 34ull * 1024 * 1024 + 256ull * 1024);  // [64][2048]

    // 1) table prep
    {
        int n = EMBD * EMBD;
        cvt_f32_f16<<<(n + 255) / 256, 256, 0, stream>>>(Wfc, wfc16, n);
        n = RELPAD * HDIM;
        prep_relK <<<(n + 255) / 256, 256, 0, stream>>>(relKf, relKp);
        prep_relVT<<<(n + 255) / 256, 256, 0, stream>>>(relVf, relVT);
    }

    // 2) per-head QKV projections (z: 0=q 1=k 2=v)
    proj_kernel<<<dim3(LSEQ / 128, BATCH * HEADS, 3), 256, 0, stream>>>(
        query, key_, value, Wq, bq, Wk, bk, Wv, bv, qh, kh, vT);

    // 3) attention with relative positions
    attn_kernel<<<dim3(LSEQ / 16, HEADS, BATCH), 32, 0, stream>>>(
        qh, kh, vT, mask, relKp, relVT, X);

    // 4) final FC
    fc_kernel<<<dim3((BATCH * LSEQ) / 128, EMBD / 64), 256, 0, stream>>>(
        X, wfc16, bfc, out);
}
